// EncryptedFeedForward_83184926589623
// MI455X (gfx1250) — matrix operand order repaired and checked by reading, not compiled, not run on hardware
//
#include <hip/hip_runtime.h>
#include <stdint.h>

// ---------------------------------------------------------------------------
// out = poly_relu(x @ W1 + b1) @ W2 + b2
//   x [8192,1024] f32, W1 [1024,4096] f32, b1[4096], W2 [4096,1024] f32,
//   b2[1024], out [8192,1024] f32.
// bf16 WMMA (v_wmma_f32_16x16x32_bf16) with fp32 accumulation;
// tile fill via GLOBAL_LOAD_ASYNC_TO_LDS_B128 (ASYNCcnt), 3-stage pipeline,
// branch-free steady state.
// ---------------------------------------------------------------------------

typedef __bf16 bf16;
typedef __bf16 v16bf __attribute__((ext_vector_type(16)));
typedef __bf16 v8bf  __attribute__((ext_vector_type(8)));
typedef __bf16 v4bf  __attribute__((ext_vector_type(4)));
typedef float  v8f   __attribute__((ext_vector_type(8)));
typedef float  v4f   __attribute__((ext_vector_type(4)));

#define NROWS 8192
#define DDIM  1024
#define FDIM  4096

// ---------------------------------------------------------------------------
__global__ void cvt_f32_bf16(const float* __restrict__ src,
                             bf16* __restrict__ dst, int n4) {
  int i = blockIdx.x * blockDim.x + threadIdx.x;
  if (i < n4) {
    v4f f = ((const v4f*)src)[i];
    v4bf o;
    o[0] = (bf16)f[0];
    o[1] = (bf16)f[1];
    o[2] = (bf16)f[2];
    o[3] = (bf16)f[3];
    ((v4bf*)dst)[i] = o;
  }
}

// Transpose + convert: src f32 [R][C] -> dst bf16 [C][R]
__global__ void tconv_f32_bf16(const float* __restrict__ src,
                               bf16* __restrict__ dst, int R, int C) {
  __shared__ float tile[32][33];
  const int tx = threadIdx.x, ty = threadIdx.y;
  const int x  = blockIdx.x * 32 + tx;
  const int y0 = blockIdx.y * 32;
  #pragma unroll
  for (int j = ty; j < 32; j += 8)
    tile[j][tx] = src[(size_t)(y0 + j) * C + x];
  __syncthreads();
  const int xo  = y0 + tx;
  const int yo0 = blockIdx.x * 32;
  #pragma unroll
  for (int j = ty; j < 32; j += 8)
    dst[(size_t)(yo0 + j) * R + xo] = (bf16)tile[tx][j];
}

// ---------------------------------------------------------------------------
// Tiled bf16 GEMM: C = A[M,K] * B^T[N,K] (+bias, +epilogue)
//   mode 0: poly_relu(acc + bias) -> bf16   mode 1: acc + bias -> f32
// 256 threads = 8 wave32; block tile 128x128, BK=32; wave = 32x64 patch
// = 2x4 v_wmma_f32_16x16x32_bf16 per K-step.
// ---------------------------------------------------------------------------
__global__ __launch_bounds__(256) void gemm_bf16_wmma(
    const bf16* __restrict__ A, const bf16* __restrict__ B,
    const float* __restrict__ bias, void* __restrict__ Cout,
    int K, int ldc, int mode) {
  constexpr int BK = 32;
  constexpr int LSTRIDE = 80;                   // padded bytes per LDS row
  constexpr int TILE_BYTES = 128 * LSTRIDE;     // 10240 B per operand tile
  constexpr int STAGE = 2 * TILE_BYTES;         // A tile + B tile
  __shared__ char smem[3 * STAGE];              // 60 KB, 3-stage pipeline

  const int tid  = threadIdx.x;
  const int lane = tid & 31;
  const int wid  = tid >> 5;
  const int lr   = lane & 15;
  const int hi   = lane >> 4;
  const int wm   = (wid & 3) * 32;
  const int wn   = (wid >> 2) * 64;

  const size_t mBase = (size_t)blockIdx.y * 128;
  const size_t nBase = (size_t)blockIdx.x * 128;
  const size_t ldab  = (size_t)K * 2;           // row bytes of A and B^T

  // Fill plan: 512 x 16B chunks per operand tile; 2 chunks/thread/operand.
  const int ra0 = tid >> 2,          ka0 = (tid & 3) * 16;
  const int ra1 = (tid + 256) >> 2,  ka1 = ((tid + 256) & 3) * 16;

  const char* gA = (const char*)A + mBase * ldab;
  const char* gB = (const char*)B + nBase * ldab;

  // Per-thread LDS byte offsets (stage 0) and running global addresses.
  const uint32_t lA0 = (uint32_t)(uintptr_t)(smem + ra0 * LSTRIDE + ka0);
  const uint32_t lA1 = (uint32_t)(uintptr_t)(smem + ra1 * LSTRIDE + ka1);
  const uint32_t lB0 = lA0 + TILE_BYTES;
  const uint32_t lB1 = lA1 + TILE_BYTES;
  uint64_t ga0 = (uint64_t)(uintptr_t)(gA + (size_t)ra0 * ldab + ka0);
  uint64_t ga1 = (uint64_t)(uintptr_t)(gA + (size_t)ra1 * ldab + ka1);
  uint64_t gb0 = (uint64_t)(uintptr_t)(gB + (size_t)ra0 * ldab + ka0);
  uint64_t gb1 = (uint64_t)(uintptr_t)(gB + (size_t)ra1 * ldab + ka1);

  // Issue one K-tile's async fills (4 instructions/wave -> ASYNCcnt += 4),
  // then advance the global cursors by BK*2 = 64 bytes.
  auto issue_tile = [&](int stage) {
    const uint32_t off = (uint32_t)stage * STAGE;
    asm volatile("global_load_async_to_lds_b128 %0, %1, off"
                 :: "v"(lA0 + off), "v"(ga0) : "memory");
    asm volatile("global_load_async_to_lds_b128 %0, %1, off"
                 :: "v"(lA1 + off), "v"(ga1) : "memory");
    asm volatile("global_load_async_to_lds_b128 %0, %1, off"
                 :: "v"(lB0 + off), "v"(gb0) : "memory");
    asm volatile("global_load_async_to_lds_b128 %0, %1, off"
                 :: "v"(lB1 + off), "v"(gb1) : "memory");
    ga0 += BK * 2; ga1 += BK * 2; gb0 += BK * 2; gb1 += BK * 2;
  };

  v8f acc[2][4];
  #pragma unroll
  for (int i = 0; i < 2; ++i)
    #pragma unroll
    for (int j = 0; j < 4; ++j)
      acc[i][j] = (v8f){};

  // Consume one staged K-tile: 12 ds_load_b128 -> 8 WMMAs.
  auto compute_tile = [&](int stage) {
    const char* as = smem + stage * STAGE;
    const char* bs = as + TILE_BYTES;
    union Frag { v16bf v; v8bf h[2]; };
    Frag aF[2], bF[4];
    #pragma unroll
    for (int mt = 0; mt < 2; ++mt) {
      const char* p = as + (wm + mt * 16 + lr) * LSTRIDE + hi * 16;
      aF[mt].h[0] = *(const v8bf*)p;
      aF[mt].h[1] = *(const v8bf*)(p + 32);
    }
    #pragma unroll
    for (int nt = 0; nt < 4; ++nt) {
      const char* p = bs + (wn + nt * 16 + lr) * LSTRIDE + hi * 16;
      bF[nt].h[0] = *(const v8bf*)p;
      bF[nt].h[1] = *(const v8bf*)(p + 32);
    }
    #pragma unroll
    for (int mt = 0; mt < 2; ++mt)
      #pragma unroll
      for (int nt = 0; nt < 4; ++nt)
        acc[mt][nt] = __builtin_amdgcn_wmma_f32_16x16x32_bf16(
            false, aF[mt].v, false, bF[nt].v, (short)0, acc[mt][nt],
            false, false);
  };

  const int nk = K / BK;   // >= 32 for both layers

  // Prologue: two tiles in flight.
  issue_tile(0);
  issue_tile(1);

  // Steady state (branch-free body): wait own fills of the read stage
  // (in-order completion: <=4 outstanding => read-stage tile landed),
  // join all waves, refill the stage read in the previous iteration, compute.
  int rs = 0, ws = 2;
  for (int kt = 0; kt < nk - 2; ++kt) {
    asm volatile("s_wait_asynccnt 0x4" ::: "memory");
    __syncthreads();
    issue_tile(ws);
    compute_tile(rs);
    rs = (rs == 2) ? 0 : rs + 1;
    ws = (ws == 2) ? 0 : ws + 1;
  }
  // Drain: two staged tiles remain.
  asm volatile("s_wait_asynccnt 0x4" ::: "memory");
  __syncthreads();
  compute_tile(rs);
  rs = (rs == 2) ? 0 : rs + 1;
  asm volatile("s_wait_asynccnt 0x0" ::: "memory");
  __syncthreads();
  compute_tile(rs);

  // Epilogue. C/D layout: VGPR r -> m = r + 8*(lane>>4), n = lane&15.
  if (mode == 0) {
    bf16* H = (bf16*)Cout;
    #pragma unroll
    for (int mt = 0; mt < 2; ++mt) {
      const size_t m0 = mBase + wm + mt * 16 + hi * 8;
      #pragma unroll
      for (int nt = 0; nt < 4; ++nt) {
        const size_t n = nBase + wn + nt * 16 + lr;
        const float bv = bias[n];
        #pragma unroll
        for (int r = 0; r < 8; ++r) {
          float v = acc[mt][nt][r] + bv;
          v = v * (0.5f + v * (0.5f - 0.125f * v));   // poly_relu
          H[(m0 + r) * (size_t)ldc + n] = (bf16)v;
        }
      }
    }
  } else {
    float* O = (float*)Cout;
    #pragma unroll
    for (int mt = 0; mt < 2; ++mt) {
      const size_t m0 = mBase + wm + mt * 16 + hi * 8;
      #pragma unroll
      for (int nt = 0; nt < 4; ++nt) {
        const size_t n = nBase + wn + nt * 16 + lr;
        const float bv = bias[n];
        #pragma unroll
        for (int r = 0; r < 8; ++r)
          O[(m0 + r) * (size_t)ldc + n] = acc[mt][nt][r] + bv;
      }
    }
  }
}

// ---------------------------------------------------------------------------
extern "C" void kernel_launch(void* const* d_in, const int* in_sizes, int n_in,
                              void* d_out, int out_size, void* d_ws,
                              size_t ws_size, hipStream_t stream) {
  const float* x  = (const float*)d_in[0];
  const float* W1 = (const float*)d_in[1];
  const float* b1 = (const float*)d_in[2];
  const float* W2 = (const float*)d_in[3];
  const float* b2 = (const float*)d_in[4];
  float* out = (float*)d_out;

  char* ws = (char*)d_ws;
  const size_t xb_bytes = (size_t)NROWS * DDIM * 2;   // 16 MB
  const size_t w_bytes  = (size_t)DDIM * FDIM * 2;    //  8 MB each
  bf16* xb  = (bf16*)(ws);
  bf16* w1t = (bf16*)(ws + xb_bytes);
  bf16* w2t = (bf16*)(ws + xb_bytes + w_bytes);
  bf16* hb  = (bf16*)(ws + xb_bytes + 2 * w_bytes);   // 64 MB

  {
    const int n4 = NROWS * DDIM / 4;
    cvt_f32_bf16<<<(n4 + 255) / 256, 256, 0, stream>>>(x, xb, n4);
  }
  tconv_f32_bf16<<<dim3(FDIM / 32, DDIM / 32), dim3(32, 8), 0, stream>>>(
      W1, w1t, DDIM, FDIM);
  tconv_f32_bf16<<<dim3(DDIM / 32, FDIM / 32), dim3(32, 8), 0, stream>>>(
      W2, w2t, FDIM, DDIM);

  // Layer 1: h = poly_relu(x @ W1 + b1) -> bf16 [8192,4096]
  gemm_bf16_wmma<<<dim3(FDIM / 128, NROWS / 128), 256, 0, stream>>>(
      xb, w1t, b1, hb, DDIM, FDIM, 0);
  // Layer 2: out = h @ W2 + b2 -> f32 [8192,1024]
  gemm_bf16_wmma<<<dim3(DDIM / 128, NROWS / 128), 256, 0, stream>>>(
      hb, w2t, b2, out, FDIM, DDIM, 1);
}
